// ClsPointTransformer_395136991310
// MI455X (gfx1250) — compile-verified, matrix operand order checked by reading, not run on hardware
//
#include <hip/hip_runtime.h>
#include <stdint.h>

// ---------------------------------------------------------------------------
// Problem constants
// ---------------------------------------------------------------------------
#define PT_B 8
#define PT_N 4096
#define PT_K 16
#define PT_D 128
#define PT_L 2
#define PT_CLS 40
#define BN (PT_B * PT_N)          // 32768 total points

typedef __attribute__((ext_vector_type(16))) _Float16 v16h;
typedef __attribute__((ext_vector_type(8)))  _Float16 v8h;
typedef __attribute__((ext_vector_type(8)))  float    v8f;
typedef __attribute__((ext_vector_type(4)))  unsigned int u32x4;
typedef __attribute__((ext_vector_type(8)))  int i32x8;
typedef __attribute__((ext_vector_type(4)))  int i32x4;

union FragA { v16h v; _Float16 e[16]; };
union Frag8 { v8h  v; _Float16 e[8];  };
union FragC { v8f  v; float    e[8];  };

// ---------------------------------------------------------------------------
// Swizzled tile layout.
// A 16-row x 128-col f16 tile is stored as 4 k-chunks x 32 lanes x 16 halfs,
// exactly the WMMA A/B fragment order (ISA 7.12.2), so a fragment load is one
// contiguous 32-byte vector LDS/global load per lane.
//   element (row, d):  kc=d/32, o=d%32, hi=(o>>3)&1, i=(o&7)+((o&16)?8:0)
//   offset = ((tile*4+kc)*32 + hi*16 + row%16)*16 + i      (tile = row/16)
// ---------------------------------------------------------------------------

__device__ __forceinline__ v8f wmma16(v16h a, v16h b, v8f c) {
  return __builtin_amdgcn_wmma_f32_16x16x32_f16(
      false, a, false, b, (short)0, c, false, false);
}

// fragment load from swizzled storage (A from activations, B from weights)
__device__ __forceinline__ v16h load_frag(const _Float16* base, int tile, int kc) {
  int lane = threadIdx.x & 31;
  return *(const v16h*)(base + ((tile * 4 + kc) * 32 + lane) * 16);
}

// store C/D fragment (tile mt, n-tile nt) into swizzled f16 storage
template <bool RELU>
__device__ __forceinline__ void store_c_sw(_Float16* buf, int mt, int nt, v8f c) {
  int lane = threadIdx.x & 31;
  int r = lane & 15, hi = lane >> 4;
  int kc2 = nt >> 1;
  int o = ((nt & 1) << 4) + r;
  int h2 = (o >> 3) & 1;
  int i2 = (o & 7) + ((o & 16) ? 8 : 0);
  _Float16* p = buf + ((mt * 4 + kc2) * 32 + h2 * 16) * 16 + i2;
  FragC f; f.v = c;
#pragma unroll
  for (int j = 0; j < 8; ++j) {
    float val = f.e[j];
    if (RELU) val = fmaxf(val, 0.f);
    p[(hi * 8 + j) * 16] = (_Float16)val;   // row = hi*8+j
  }
}

// 256x128 = 256x128 @ 128x128, all operands swizzled, 8 waves
template <bool RELU>
__device__ __forceinline__ void gemm_sw(const _Float16* A, const _Float16* W,
                                        _Float16* Dst) {
  int wave = threadIdx.x >> 5;
#pragma unroll
  for (int mi = 0; mi < 2; ++mi) {
    int mt = wave * 2 + mi;
    v16h a0 = load_frag(A, mt, 0);
    v16h a1 = load_frag(A, mt, 1);
    v16h a2 = load_frag(A, mt, 2);
    v16h a3 = load_frag(A, mt, 3);
#pragma unroll
    for (int nt = 0; nt < 8; ++nt) {
      v8f c = {};
      c = wmma16(a0, load_frag(W, nt, 0), c);
      c = wmma16(a1, load_frag(W, nt, 1), c);
      c = wmma16(a2, load_frag(W, nt, 2), c);
      c = wmma16(a3, load_frag(W, nt, 3), c);
      store_c_sw<RELU>(Dst, mt, nt, c);
    }
  }
}

// ---------------------------------------------------------------------------
// Tensor Data Mover: DMA a contiguous run of dwords from global into LDS.
// D# built per ISA 8.3/8.4: group0 {count=1, lds_addr, global_addr, type=2},
// group1 {data_size=4B, tensor_dim0=nd, tensor_dim1=1, tile_dim0=nd,
//         tile_dim1=1, dim0_stride=nd}. Tracked by TENSORcnt.
// ---------------------------------------------------------------------------
__device__ __forceinline__ void tdm_load_to_lds(void* ldsDst, const void* gsrc, int nd) {
  unsigned int ldsOff = (unsigned int)(uintptr_t)ldsDst;   // low 32 bits = LDS offset
  unsigned long long ga = (unsigned long long)(uintptr_t)gsrc;
  u32x4 g0 = { 1u, ldsOff, (unsigned int)ga,
               (unsigned int)(ga >> 32) | (2u << 30) };
  i32x8 g1 = { (int)(2u << 16),                       // data_size = 4 bytes
               (int)(((unsigned)nd & 0xffffu) << 16), // tensor_dim0 lo
               (int)((((unsigned)nd >> 16) & 0xffffu) | (1u << 16)), // dim0 hi | dim1=1
               (int)(((unsigned)nd & 0xffffu) << 16), // tile_dim0
               1,                                     // tile_dim1 = 1
               nd, 0, 0 };                            // dim0_stride
  i32x4 gz = { 0, 0, 0, 0 };
#if defined(__clang_major__) && __clang_major__ >= 23
  i32x8 gz8 = { 0, 0, 0, 0, 0, 0, 0, 0 };
  __builtin_amdgcn_tensor_load_to_lds(g0, g1, gz, gz, gz8, 0);
#else
  __builtin_amdgcn_tensor_load_to_lds(g0, g1, gz, gz, 0);
#endif
}

// stage one 128x128 f16 weight (already swizzled) into LDS: each of 8 waves
// DMAs a 4KB slice, then waits on its own TENSORcnt (caller barriers after).
__device__ __forceinline__ void stage_weight_tdm(_Float16* wT, const _Float16* wsrc) {
  int wave = threadIdx.x >> 5;
  tdm_load_to_lds(wT + wave * 2048, wsrc + wave * 2048, 1024);
  __builtin_amdgcn_s_wait_tensorcnt(0);
}

// ---------------------------------------------------------------------------
// Kernel: f32 -> f16 weight conversion into swizzled fragment order
// ---------------------------------------------------------------------------
__global__ void cvt_sw_kernel(const float* __restrict__ s, _Float16* __restrict__ d,
                              int nmat) {
  int os = blockIdx.x * 256 + threadIdx.x;           // one output half per thread
  int mat = os >> 14;
  if (mat >= nmat) return;
  int t = os & 16383;
  int i = t & 15, lane = (t >> 4) & 31, kc = (t >> 9) & 3, nt = t >> 11;
  int r = lane & 15, hi = lane >> 4;
  int kdim = kc * 32 + hi * 8 + (i & 7) + ((i & 8) ? 16 : 0);
  int col = nt * 16 + r;
  d[os] = (_Float16)s[mat * 16384 + kdim * 128 + col];
}

// ---------------------------------------------------------------------------
// Kernel: embedding  x = relu(features @ W_embed); writes x (f32 linear) and
// xh (f16, globally swizzled in fragment order). One thread per 16-half group.
// ---------------------------------------------------------------------------
__global__ void embed_kernel(const float* __restrict__ f, const float* __restrict__ We,
                             float* __restrict__ x, _Float16* __restrict__ xh) {
  int g = blockIdx.x * 256 + threadIdx.x;            // BN*128/16 groups
  int lane = g & 31, kc = (g >> 5) & 3;
  long tile = g >> 7;
  int r = lane & 15, hi = lane >> 4;
  long row = tile * 16 + r;
  const float* fr = f + row * 3;
  float f0 = fr[0], f1 = fr[1], f2 = fr[2];
  int dbase = kc * 32 + hi * 8;
  FragA o;
  float tmp[16];
#pragma unroll
  for (int i = 0; i < 16; ++i) {
    int d = dbase + (i & 7) + ((i & 8) ? 16 : 0);
    float a = fmaxf(f0 * We[d] + f1 * We[128 + d] + f2 * We[256 + d], 0.f);
    o.e[i] = (_Float16)a;
    tmp[i] = a;
  }
  *(v16h*)(xh + (long)g * 16) = o.v;
  float* xr = x + row * 128 + dbase;                 // two contiguous 8-float runs
#pragma unroll
  for (int i = 0; i < 8; ++i) { xr[i] = tmp[i]; xr[16 + i] = tmp[i + 8]; }
}

// ---------------------------------------------------------------------------
// Kernel: kNN (top-16 nearest incl. self), 1 thread per query, LDS-tiled
// ---------------------------------------------------------------------------
__global__ void knn_kernel(const float* __restrict__ pos, int* __restrict__ idx) {
  __shared__ float pt[256 * 3];
  int g = blockIdx.x * 256 + threadIdx.x;
  int b = g >> 12;
  long base = (long)b * PT_N;
  float px = pos[(long)g * 3], py = pos[(long)g * 3 + 1], pz = pos[(long)g * 3 + 2];
  float bd[16]; int bi[16];
#pragma unroll
  for (int s = 0; s < 16; ++s) { bd[s] = 1e30f; bi[s] = 0; }
  for (int j0 = 0; j0 < PT_N; j0 += 256) {
    __syncthreads();
    int t = threadIdx.x;
    pt[t * 3]     = pos[(base + j0 + t) * 3];
    pt[t * 3 + 1] = pos[(base + j0 + t) * 3 + 1];
    pt[t * 3 + 2] = pos[(base + j0 + t) * 3 + 2];
    __syncthreads();
    for (int j = 0; j < 256; ++j) {
      float dx = px - pt[j * 3], dy = py - pt[j * 3 + 1], dz = pz - pt[j * 3 + 2];
      float d2 = dx * dx + dy * dy + dz * dz;
      if (d2 < bd[15]) {
        float cd = d2; int ci = j0 + j;
#pragma unroll
        for (int s = 0; s < 16; ++s) {
          if (cd < bd[s]) {
            float td = bd[s]; int ti = bi[s];
            bd[s] = cd; bi[s] = ci; cd = td; ci = ti;
          }
        }
      }
    }
  }
#pragma unroll
  for (int s = 0; s < 16; ++s) idx[(long)g * 16 + s] = bi[s];
}

// ---------------------------------------------------------------------------
// Kernel: QKV projections. A fragments come straight from swizzled xh in L2;
// weights staged into LDS via TDM. 1 WG = 8 tiles (128 rows).
// ---------------------------------------------------------------------------
__global__ void qkv_kernel(const _Float16* __restrict__ xh,
                           const _Float16* __restrict__ wq,
                           const _Float16* __restrict__ wk,
                           const _Float16* __restrict__ wv,
                           float* __restrict__ q, float* __restrict__ kk,
                           float* __restrict__ v) {
  extern __shared__ char smem[];
  _Float16* wT = (_Float16*)smem;                    // 32 KB
  int wave = threadIdx.x >> 5;
  int lane = threadIdx.x & 31;
  long mtile = (long)blockIdx.x * 8 + wave;
  const _Float16* asrc = xh + mtile * 2048 + lane * 16;
  v16h a0 = *(const v16h*)(asrc);
  v16h a1 = *(const v16h*)(asrc + 512);
  v16h a2 = *(const v16h*)(asrc + 1024);
  v16h a3 = *(const v16h*)(asrc + 1536);
  const _Float16* wptr[3] = { wq, wk, wv };
  float* optr[3] = { q, kk, v };
  int r = lane & 15, hi = lane >> 4;
  for (int s = 0; s < 3; ++s) {
    __syncthreads();
    stage_weight_tdm(wT, wptr[s]);
    __syncthreads();
#pragma unroll
    for (int nt = 0; nt < 8; ++nt) {
      v8f c = {};
      c = wmma16(a0, load_frag(wT, nt, 0), c);
      c = wmma16(a1, load_frag(wT, nt, 1), c);
      c = wmma16(a2, load_frag(wT, nt, 2), c);
      c = wmma16(a3, load_frag(wT, nt, 3), c);
      FragC f; f.v = c;
#pragma unroll
      for (int j = 0; j < 8; ++j)
        optr[s][(mtile * 16 + hi * 8 + j) * 128 + nt * 16 + r] = f.e[j];
    }
  }
}

// ---------------------------------------------------------------------------
// Kernel: fused point-transformer block for a tile of 16 queries.
// All [256x128] intermediates live in LDS in swizzled fragment order.
// ---------------------------------------------------------------------------
__global__ void attn_kernel(float* __restrict__ x, _Float16* __restrict__ xh,
                            const float* __restrict__ q, const float* __restrict__ kk,
                            const float* __restrict__ v, const int* __restrict__ idx,
                            const float* __restrict__ pos, const float* __restrict__ P1,
                            const _Float16* __restrict__ P2h,
                            const _Float16* __restrict__ G1h,
                            const _Float16* __restrict__ G2h,
                            const _Float16* __restrict__ Woh) {
  extern __shared__ char smem[];
  _Float16* buf0  = (_Float16*)smem;            // 256x128 (hidden -> g -> logits)
  _Float16* buf1  = buf0 + 32768;               // 256x128 (delta, live to end)
  _Float16* buf2  = buf1 + 32768;               // 256x128 (h1)
  _Float16* wbuf  = buf2 + 32768;               // 128x128 weight
  float*    qLds  = (float*)(wbuf + 16384);     // 16x128 f32 (row-major)
  float*    relLds = qLds + 2048;               // 256x3 f32
  float*    p1Lds  = relLds + 768;              // 3x128 f32
  _Float16* aggLds = (_Float16*)(p1Lds + 384);  // 16x128 swizzled (1 tile)
  int*      idxLds = (int*)(aggLds + 2048);     // 256 int

  int tid = threadIdx.x;
  int gq0 = blockIdx.x * 16;
  int b = gq0 >> 12;
  int n0 = gq0 & (PT_N - 1);
  long rowBase = (long)b * PT_N;

  // stage neighbor indices + relative positions (one (qi,k) per thread)
  {
    int qi = tid >> 4, k = tid & 15;
    int nbr = idx[(rowBase + n0 + qi) * PT_K + k];
    idxLds[tid] = nbr;
    const float* pi = pos + (rowBase + n0 + qi) * 3;
    const float* pj = pos + (rowBase + nbr) * 3;
    relLds[tid * 3]     = pi[0] - pj[0];
    relLds[tid * 3 + 1] = pi[1] - pj[1];
    relLds[tid * 3 + 2] = pi[2] - pj[2];
  }
  for (int e = tid; e < 2048; e += 256)
    qLds[e] = q[(rowBase + n0 + (e >> 7)) * 128 + (e & 127)];
  for (int e = tid; e < 384; e += 256) p1Lds[e] = P1[e];
  __syncthreads();

  // hidden = relu(rel @ P1) -> buf0 (iterate directly in swizzled group order)
  for (int g = tid; g < 2048; g += 256) {
    int lane = g & 31, kc = (g >> 5) & 3, mt = g >> 7;
    int r = lane & 15, hi = lane >> 4;
    int row = mt * 16 + r;
    float r0 = relLds[row * 3], r1 = relLds[row * 3 + 1], r2 = relLds[row * 3 + 2];
    int dbase = kc * 32 + hi * 8;
    FragA o;
#pragma unroll
    for (int i = 0; i < 16; ++i) {
      int d = dbase + (i & 7) + ((i & 8) ? 16 : 0);
      o.e[i] = (_Float16)fmaxf(r0 * p1Lds[d] + r1 * p1Lds[128 + d] + r2 * p1Lds[256 + d],
                               0.f);
    }
    *(v16h*)(buf0 + g * 16) = o.v;
  }
  stage_weight_tdm(wbuf, P2h);
  __syncthreads();
  gemm_sw<false>(buf0, wbuf, buf1);             // delta -> buf1
  __syncthreads();

  // g = q_i - k_j + delta -> buf0
  for (int g = tid; g < 2048; g += 256) {
    int lane = g & 31, kc = (g >> 5) & 3, mt = g >> 7;
    int r = lane & 15, hi = lane >> 4;
    int row = mt * 16 + r;                      // qi = mt, k = r
    int nbr = idxLds[row];
    const float* krow = kk + (rowBase + nbr) * 128;
    const float* qrow = qLds + mt * 128;
    FragA dl; dl.v = *(const v16h*)(buf1 + g * 16);
    int dbase = kc * 32 + hi * 8;
    FragA o;
#pragma unroll
    for (int i = 0; i < 16; ++i) {
      int d = dbase + (i & 7) + ((i & 8) ? 16 : 0);
      o.e[i] = (_Float16)(qrow[d] - krow[d] + (float)dl.e[i]);
    }
    *(v16h*)(buf0 + g * 16) = o.v;
  }
  stage_weight_tdm(wbuf, G1h);
  __syncthreads();
  gemm_sw<true>(buf0, wbuf, buf2);              // h1 = relu(g@G1) -> buf2
  __syncthreads();
  stage_weight_tdm(wbuf, G2h);
  __syncthreads();
  gemm_sw<false>(buf2, wbuf, buf0);             // logits -> buf0
  __syncthreads();

  // channel-wise softmax over K + aggregation.
  // thread -> (qi, kc, h2, half): 8 channels x 16 neighbors, vector LDS loads.
  {
    int qi = tid >> 4, kc = (tid >> 2) & 3, h2 = (tid >> 1) & 1, hf = tid & 1;
    const _Float16* lbase = buf0 + ((qi * 4 + kc) * 32 + h2 * 16) * 16 + hf * 8;
    const _Float16* dbase = buf1 + ((qi * 4 + kc) * 32 + h2 * 16) * 16 + hf * 8;
    int d0 = kc * 32 + h2 * 8 + hf * 16;
    float m[8], ssum[8], acc[8];
#pragma unroll
    for (int u = 0; u < 8; ++u) { m[u] = -1e30f; ssum[u] = 0.f; acc[u] = 0.f; }
#pragma unroll
    for (int k = 0; k < 16; ++k) {
      Frag8 lv; lv.v = *(const v8h*)(lbase + k * 16);
#pragma unroll
      for (int u = 0; u < 8; ++u) m[u] = fmaxf(m[u], (float)lv.e[u]);
    }
#pragma unroll
    for (int k = 0; k < 16; ++k) {
      Frag8 lv; lv.v = *(const v8h*)(lbase + k * 16);
      Frag8 dv; dv.v = *(const v8h*)(dbase + k * 16);
      int nbr = idxLds[qi * 16 + k];
      const float* vrow = v + (rowBase + nbr) * 128 + d0;
#pragma unroll
      for (int u = 0; u < 8; ++u) {
        float w = __expf((float)lv.e[u] - m[u]);
        ssum[u] += w;
        acc[u] += w * (vrow[u] + (float)dv.e[u]);
      }
    }
    Frag8 o;
#pragma unroll
    for (int u = 0; u < 8; ++u) o.e[u] = (_Float16)(acc[u] / ssum[u]);
    *(v8h*)(aggLds + (kc * 32 + h2 * 16 + qi) * 16 + hf * 8) = o.v;
  }
  stage_weight_tdm(wbuf, Woh);
  __syncthreads();

  // out = agg @ Wo (M=16), residual add, write x (f32) + xh (swizzled f16)
  {
    int wave = tid >> 5, nt = wave;
    v16h a0 = load_frag(aggLds, 0, 0);
    v16h a1 = load_frag(aggLds, 0, 1);
    v16h a2 = load_frag(aggLds, 0, 2);
    v16h a3 = load_frag(aggLds, 0, 3);
    v8f c = {};
    c = wmma16(a0, load_frag(wbuf, nt, 0), c);
    c = wmma16(a1, load_frag(wbuf, nt, 1), c);
    c = wmma16(a2, load_frag(wbuf, nt, 2), c);
    c = wmma16(a3, load_frag(wbuf, nt, 3), c);
    int lane = tid & 31, r = lane & 15, hi = lane >> 4;
    int kc2 = nt >> 1, o = ((nt & 1) << 4) + r;
    int h2 = (o >> 3) & 1, i2 = (o & 7) + ((o & 16) ? 8 : 0);
    long tile = (rowBase + n0) >> 4;
    _Float16* xhp = xh + tile * 2048 + (kc2 * 32 + h2 * 16) * 16 + i2;
    FragC f; f.v = c;
#pragma unroll
    for (int j = 0; j < 8; ++j) {
      int mrow = hi * 8 + j;
      long off = (rowBase + n0 + mrow) * 128 + nt * 16 + r;
      float xo = x[off] + f.e[j];
      x[off] = xo;
      xhp[mrow * 16] = (_Float16)xo;
    }
  }
}

// ---------------------------------------------------------------------------
// Kernel: global max pool over N
// ---------------------------------------------------------------------------
__global__ void pool_kernel(const float* __restrict__ x, float* __restrict__ pooled) {
  int b = blockIdx.x, d = threadIdx.x;               // 8 blocks x 128 threads
  float m = -1e30f;
  const float* xp = x + (long)b * PT_N * 128 + d;
  for (int n = 0; n < PT_N; ++n) m = fmaxf(m, xp[(long)n * 128]);
  pooled[b * 128 + d] = m;
}

// ---------------------------------------------------------------------------
// Kernel: classifier head (tiny)
// ---------------------------------------------------------------------------
__global__ void cls_kernel(const float* __restrict__ pooled, const float* __restrict__ W1,
                           const float* __restrict__ W2, float* __restrict__ out) {
  __shared__ float h[PT_B * 128];
  for (int e = threadIdx.x; e < PT_B * 128; e += 256) {
    int b = e >> 7, hd = e & 127;
    float a = 0.f;
    for (int dd = 0; dd < 128; ++dd) a += pooled[b * 128 + dd] * W1[dd * 128 + hd];
    h[e] = fmaxf(a, 0.f);
  }
  __syncthreads();
  for (int e = threadIdx.x; e < PT_B * PT_CLS; e += 256) {
    int b = e / PT_CLS, c = e % PT_CLS;
    float a = 0.f;
    for (int dd = 0; dd < 128; ++dd) a += h[b * 128 + dd] * W2[dd * PT_CLS + c];
    out[e] = a;
  }
}

// ---------------------------------------------------------------------------
// Host launcher
// ---------------------------------------------------------------------------
extern "C" void kernel_launch(void* const* d_in, const int* in_sizes, int n_in,
                              void* d_out, int out_size, void* d_ws, size_t ws_size,
                              hipStream_t stream) {
  const float* features = (const float*)d_in[0];
  const float* pos      = (const float*)d_in[1];
  const float* W_embed  = (const float*)d_in[2];
  const float* Wq       = (const float*)d_in[3];
  const float* Wk       = (const float*)d_in[4];
  const float* Wv       = (const float*)d_in[5];
  const float* P1       = (const float*)d_in[6];
  const float* P2       = (const float*)d_in[7];
  const float* G1       = (const float*)d_in[8];
  const float* G2       = (const float*)d_in[9];
  const float* Wo       = (const float*)d_in[10];
  const float* W_cls1   = (const float*)d_in[11];
  const float* W_cls2   = (const float*)d_in[12];
  float* out = (float*)d_out;

  // ---- workspace layout (~75 MB) ----
  char* ws = (char*)d_ws;
  const size_t XN = (size_t)BN * 128;
  float*    x    = (float*)(ws);                       // 16 MB (f32, linear)
  _Float16* xh   = (_Float16*)(ws + XN * 4);           //  8 MB (f16, swizzled)
  int*      idxb = (int*)(ws + XN * 6);                //  2 MB
  float*    qb   = (float*)(ws + XN * 6 + (size_t)BN * PT_K * 4);
  float*    kb   = qb + XN;                            // 16 MB each
  float*    vb   = kb + XN;
  _Float16* wqh  = (_Float16*)(vb + XN);               // 7 x (2*128*128) f16 swizzled
  _Float16* wkh  = wqh + 2 * 16384;
  _Float16* wvh  = wkh + 2 * 16384;
  _Float16* p2h  = wvh + 2 * 16384;
  _Float16* g1h  = p2h + 2 * 16384;
  _Float16* g2h  = g1h + 2 * 16384;
  _Float16* woh  = g2h + 2 * 16384;
  float*    pooled = (float*)(woh + 2 * 16384);        // 8x128

  // weight conversion + swizzle (2 layer matrices each)
  cvt_sw_kernel<<<128, 256, 0, stream>>>(Wq, wqh, 2);
  cvt_sw_kernel<<<128, 256, 0, stream>>>(Wk, wkh, 2);
  cvt_sw_kernel<<<128, 256, 0, stream>>>(Wv, wvh, 2);
  cvt_sw_kernel<<<128, 256, 0, stream>>>(P2, p2h, 2);
  cvt_sw_kernel<<<128, 256, 0, stream>>>(G1, g1h, 2);
  cvt_sw_kernel<<<128, 256, 0, stream>>>(G2, g2h, 2);
  cvt_sw_kernel<<<128, 256, 0, stream>>>(Wo, woh, 2);

  embed_kernel<<<(int)(XN / 16 / 256), 256, 0, stream>>>(features, W_embed, x, xh);
  knn_kernel<<<BN / 256, 256, 0, stream>>>(pos, idxb);

  const size_t qkv_smem  = 16384 * sizeof(_Float16);   // 32 KB weight buffer
  const size_t attn_smem = 3 * 32768 * sizeof(_Float16)
                         + 16384 * sizeof(_Float16)
                         + 2048 * 4 + 768 * 4 + 384 * 4
                         + 2048 * sizeof(_Float16)
                         + 256 * 4;                    // ~242 KB

  for (int l = 0; l < PT_L; ++l) {
    qkv_kernel<<<BN / 128, 256, qkv_smem, stream>>>(
        xh, wqh + (size_t)l * 16384, wkh + (size_t)l * 16384,
        wvh + (size_t)l * 16384, qb, kb, vb);
    attn_kernel<<<BN / 16, 256, attn_smem, stream>>>(
        x, xh, qb, kb, vb, idxb, pos, P1 + (size_t)l * 384,
        p2h + (size_t)l * 16384, g1h + (size_t)l * 16384,
        g2h + (size_t)l * 16384, woh + (size_t)l * 16384);
  }

  pool_kernel<<<PT_B, 128, 0, stream>>>(x, pooled);
  cls_kernel<<<1, 256, 0, stream>>>(pooled, W_cls1, W_cls2, out);
  (void)in_sizes; (void)n_in; (void)out_size; (void)ws_size;
}
